// AttentionHead_87411174409087
// MI455X (gfx1250) — matrix operand order repaired and checked by reading, not compiled, not run on hardware
//
#include <hip/hip_runtime.h>

typedef _Float16 v16h __attribute__((ext_vector_type(16)));
typedef _Float16 v8h  __attribute__((ext_vector_type(8)));
typedef float    v8f  __attribute__((ext_vector_type(8)));
typedef float    v4f  __attribute__((ext_vector_type(4)));

constexpr int HD   = 64;    // head size
constexpr int NEMB = 384;   // embedding dim
constexpr int SEQ  = 256;   // T
constexpr int NB   = 256;   // batch

__device__ __forceinline__ v8f wmma16(v16h a, v16h b, v8f c) {
  // D = A(16x32 f16) * B(32x16 f16) + C(16x16 f32)
  return __builtin_amdgcn_wmma_f32_16x16x32_f16(false, a, false, b, (short)0, c,
                                                false, false);
}

// ---------------------------------------------------------------------------
// Kernel 0: transpose + f16-convert the three weight matrices.
//   W[k][h] (f32, 384x64) -> Wt[w][h][k] (f16) so WMMA B-operands are
//   contiguous along the contraction (k) dimension.
// ---------------------------------------------------------------------------
__global__ void wt_xpose_kernel(const float* __restrict__ Wq,
                                const float* __restrict__ Wk,
                                const float* __restrict__ Wv,
                                _Float16* __restrict__ Wt) {
  int idx = blockIdx.x * blockDim.x + threadIdx.x;
  constexpr int total = 3 * HD * NEMB;
  if (idx >= total) return;
  int w = idx / (HD * NEMB);
  int r = idx % (HD * NEMB);
  int h = r / NEMB;
  int k = r % NEMB;
  const float* W = (w == 0) ? Wq : (w == 1) ? Wk : Wv;
  Wt[idx] = (_Float16)W[k * HD + h];
}

// ---------------------------------------------------------------------------
// Kernel 1: fused QKV projection (M=65536, K=384, N=64 x3), WMMA f16->f32.
// One wave handles a 16-row tile of all three outputs (12 accumulators).
// Writes: Qh[b][t][h] f16 (pre-scaled by 1/sqrt(64)), Kh[b][t][h] f16,
//         Vt[b][h][t] f16 (transposed for the P*V B-operand).
// ---------------------------------------------------------------------------
__global__ __launch_bounds__(128) void qkv_kernel(
    const float* __restrict__ x, const _Float16* __restrict__ Wt,
    const float* __restrict__ bq, const float* __restrict__ bk,
    const float* __restrict__ bv,
    _Float16* __restrict__ Qh, _Float16* __restrict__ Kh,
    _Float16* __restrict__ Vt) {
  int wave = threadIdx.x >> 5;
  int lane = threadIdx.x & 31;
  int hl = lane >> 4;                       // lane half (A/B K-split)
  int n  = lane & 15;                       // lane-in-half: A row / B column
  int r0 = (blockIdx.x * 4 + wave) * 16;    // first row of this wave's tile

  v8f zero = {};
  v8f acc[3][4];
#pragma unroll
  for (int w = 0; w < 3; ++w)
#pragma unroll
    for (int c = 0; c < 4; ++c) acc[w][c] = zero;

  const float* xrow = x + (size_t)(r0 + n) * NEMB;

  for (int kb = 0; kb < NEMB / 32; ++kb) {
    int kbase = kb * 32;
    // A operand: x rows r0..r0+15, k = kbase..kbase+31 (f32 -> f16).
    // A order used here (contiguous): lanes<16 hold K 0..15; lanes>=16 hold K 16..31.
    v16h a;
    {
      int kA = kbase + 16 * hl;
      v4f x0 = *(const v4f*)(xrow + kA);
      v4f x1 = *(const v4f*)(xrow + kA + 4);
      v4f x2 = *(const v4f*)(xrow + kA + 8);
      v4f x3 = *(const v4f*)(xrow + kA + 12);
#pragma unroll
      for (int e = 0; e < 4; ++e) {
        a[e]      = (_Float16)x0[e];
        a[4 + e]  = (_Float16)x1[e];
        a[8 + e]  = (_Float16)x2[e];
        a[12 + e] = (_Float16)x3[e];
      }
    }
#pragma unroll
    for (int w = 0; w < 3; ++w) {
#pragma unroll
      for (int c = 0; c < 4; ++c) {
        // B 32x16 layout: lane n = column, 16 consecutive K per lane half.
        const _Float16* bp =
            Wt + ((size_t)(w * HD + 16 * c + n) * NEMB + kbase + 16 * hl);
        v16h b16 = *(const v16h*)bp;
        acc[w][c] = wmma16(a, b16, acc[w][c]);
      }
    }
  }

  // Epilogue: bias, scale Q by 1/sqrt(HD), convert to f16, store.
#pragma unroll
  for (int c = 0; c < 4; ++c) {
    int col = 16 * c + n;
    float bqv = bq[col], bkv = bk[col], bvv = bv[col];
#pragma unroll
    for (int j = 0; j < 8; ++j) {
      size_t rr = (size_t)(r0 + j + 8 * hl);   // D layout: M = j + 8*half
      Qh[rr * HD + col] = (_Float16)((acc[0][c][j] + bqv) * 0.125f);
      Kh[rr * HD + col] = (_Float16)(acc[1][c][j] + bkv);
      int bb = (int)(rr >> 8);
      int tt = (int)(rr & 255);
      Vt[((size_t)(bb * HD + col) << 8) + tt] = (_Float16)(acc[2][c][j] + bvv);
    }
  }
}

// ---------------------------------------------------------------------------
// Kernel 2: causal flash attention. One wave per (batch, 16-row Q tile).
// s-tiles of 32 columns: 4 WMMAs for S = Q*K^T, online softmax with
// shfl_xor row reductions, P f32->f16 transposed via per-wave LDS tile,
// 4 WMMAs for O += P*V. Normalize, store fp32.
// ---------------------------------------------------------------------------
__global__ __launch_bounds__(128) void attn_kernel(
    const _Float16* __restrict__ Qh, const _Float16* __restrict__ Kh,
    const _Float16* __restrict__ Vt, float* __restrict__ out) {
  __shared__ _Float16 pbuf[4][16 * 32];   // 1 KB per wave
  int wave = threadIdx.x >> 5;
  int lane = threadIdx.x & 31;
  int hl = lane >> 4;
  int n  = lane & 15;
  int gw = blockIdx.x * 4 + wave;         // 0..4095
  int b  = gw >> 4;                       // batch
  int t0 = (gw & 15) * 16;                // first query row of tile

  const _Float16* Qb = Qh + ((size_t)b * SEQ + t0) * HD;
  const _Float16* Kb = Kh + (size_t)b * SEQ * HD;
  const _Float16* Vb = Vt + (size_t)b * HD * SEQ;

  // Q tile as two A operands (h = 0..31 and 32..63).
  v16h aQ0, aQ1;
  {
    const _Float16* qrow = Qb + n * HD;
    int kA = 16 * hl;
    v8h q00 = *(const v8h*)(qrow + kA);
    v8h q01 = *(const v8h*)(qrow + kA + 8);
    v8h q10 = *(const v8h*)(qrow + 32 + kA);
    v8h q11 = *(const v8h*)(qrow + 32 + kA + 8);
#pragma unroll
    for (int e = 0; e < 8; ++e) {
      aQ0[e] = q00[e]; aQ0[8 + e] = q01[e];
      aQ1[e] = q10[e]; aQ1[8 + e] = q11[e];
    }
  }

  v8f zero = {};
  v8f o[4] = {zero, zero, zero, zero};    // O tile 16x64 f32
  float rm[8], rl[8];                     // per-row running max / sum
#pragma unroll
  for (int j = 0; j < 8; ++j) { rm[j] = -3.0e38f; rl[j] = 0.0f; }

  _Float16* pw = &pbuf[wave][0];
  int jmax = (t0 + 15) >> 5;              // last 32-col s-tile touching diag
  for (int js = 0; js <= jmax; ++js) {
    int s0 = js * 32;
    if (js < jmax) {
      __builtin_prefetch(Kb + (size_t)(s0 + 32 + n) * HD, 0, 3);
      __builtin_prefetch(Kb + (size_t)(s0 + 48 + n) * HD, 0, 3);
    }
    // ---- S(16x32) = Q . K[s0..s0+31,:]^T (scale folded into Q) ----
    v8f S0 = zero, S1 = zero;
    {
      const _Float16* k0 = Kb + (size_t)(s0 + n) * HD + 16 * hl;
      v16h bk0 = *(const v16h*)k0;
      S0 = wmma16(aQ0, bk0, S0);
      v16h bk1 = *(const v16h*)(k0 + 32);
      S0 = wmma16(aQ1, bk1, S0);
      const _Float16* k1 = Kb + (size_t)(s0 + 16 + n) * HD + 16 * hl;
      v16h bk2 = *(const v16h*)k1;
      S1 = wmma16(aQ0, bk2, S1);
      v16h bk3 = *(const v16h*)(k1 + 32);
      S1 = wmma16(aQ1, bk3, S1);
    }
    // ---- causal mask + online softmax; D layout: row = j + 8*hl ----
#pragma unroll
    for (int j = 0; j < 8; ++j) {
      int row = t0 + j + 8 * hl;
      float v0 = (s0 + n      > row) ? -3.0e38f : S0[j];
      float v1 = (s0 + 16 + n > row) ? -3.0e38f : S1[j];
      float m = fmaxf(v0, v1);
      m = fmaxf(m, __shfl_xor(m, 1));
      m = fmaxf(m, __shfl_xor(m, 2));
      m = fmaxf(m, __shfl_xor(m, 4));
      m = fmaxf(m, __shfl_xor(m, 8));
      float nm = fmaxf(rm[j], m);
      float corr = __expf(rm[j] - nm);
      float p0 = __expf(v0 - nm);
      float p1 = __expf(v1 - nm);
      float ts = p0 + p1;
      ts += __shfl_xor(ts, 1);
      ts += __shfl_xor(ts, 2);
      ts += __shfl_xor(ts, 4);
      ts += __shfl_xor(ts, 8);
      rl[j] = rl[j] * corr + ts;
      rm[j] = nm;
#pragma unroll
      for (int c = 0; c < 4; ++c) o[c][j] *= corr;
      // stage P tile (row-major 16x32) in LDS for C->A layout transpose
      pw[(j + 8 * hl) * 32 + n]      = (_Float16)p0;
      pw[(j + 8 * hl) * 32 + 16 + n] = (_Float16)p1;
    }
    asm volatile("s_wait_dscnt 0x0" ::: "memory");
    // ---- reload P in A-operand (16x32 f16) layout ----
    v16h aP;
    {
      const _Float16* prow = pw + n * 32;
      int kA = 16 * hl;
      v8h p0v = *(const v8h*)(prow + kA);
      v8h p1v = *(const v8h*)(prow + 8 + kA);
#pragma unroll
      for (int e = 0; e < 8; ++e) { aP[e] = p0v[e]; aP[8 + e] = p1v[e]; }
    }
    // ---- O += P . V  (V pre-transposed: contiguous along s) ----
    const _Float16* vb0 = Vb + s0 + 16 * hl;
#pragma unroll
    for (int c = 0; c < 4; ++c) {
      v16h bvv = *(const v16h*)(vb0 + (size_t)(16 * c + n) * SEQ);
      o[c] = wmma16(aP, bvv, o[c]);
    }
  }

  // ---- normalize + store fp32 output ----
  float* ob = out + ((size_t)b * SEQ + t0) * HD;
#pragma unroll
  for (int j = 0; j < 8; ++j) {
    float inv = 1.0f / rl[j];
    int row = j + 8 * hl;
#pragma unroll
    for (int c = 0; c < 4; ++c)
      ob[(size_t)row * HD + 16 * c + n] = o[c][j] * inv;
  }
}

// ---------------------------------------------------------------------------
extern "C" void kernel_launch(void* const* d_in, const int* in_sizes, int n_in,
                              void* d_out, int out_size, void* d_ws,
                              size_t ws_size, hipStream_t stream) {
  const float* x  = (const float*)d_in[0];
  const float* Wq = (const float*)d_in[1];
  const float* bq = (const float*)d_in[2];
  const float* Wk = (const float*)d_in[3];
  const float* bk = (const float*)d_in[4];
  const float* Wv = (const float*)d_in[5];
  const float* bv = (const float*)d_in[6];
  float* out = (float*)d_out;

  // Workspace layout (~24.2 MB): Wt f16 | Qh f16 | Kh f16 | Vt f16
  char* ws = (char*)d_ws;
  _Float16* Wt = (_Float16*)ws;
  size_t wtBytes = ((size_t)3 * HD * NEMB * sizeof(_Float16) + 255) & ~(size_t)255;
  _Float16* Qh = (_Float16*)(ws + wtBytes);
  size_t qkvElems = (size_t)NB * SEQ * HD;
  _Float16* Kh = Qh + qkvElems;
  _Float16* Vt = Kh + qkvElems;

  wt_xpose_kernel<<<(3 * HD * NEMB + 255) / 256, 256, 0, stream>>>(Wq, Wk, Wv, Wt);
  // 65536 rows / 16 per wave / 4 waves per block = 1024 blocks
  qkv_kernel<<<(NB * SEQ / 16) / 4, 128, 0, stream>>>(x, Wt, bq, bk, bv, Qh, Kh, Vt);
  // 256 batches * 16 row-tiles / 4 waves per block = 1024 blocks
  attn_kernel<<<(NB * (SEQ / 16)) / 4, 128, 0, stream>>>(Qh, Kh, Vt, out);
}
